// ContinuousGenHyperConnectionsStrang_52939766890648
// MI455X (gfx1250) — matrix operand order repaired and checked by reading, compiled, not run on hardware
//
#include <hip/hip_runtime.h>
#include <hip/hip_bf16.h>

#define BSZ   4096
#define NN    4
#define EMBED 2048
#define DFF   8192
#define IDIM  8192
#define DT_MIN 1e-3f
#define DT_MAX 1.0f
#define REPS  1.1920929e-7f

typedef __bf16 bf16_t;
typedef __bf16 v16bf __attribute__((ext_vector_type(16)));
typedef float  v8f   __attribute__((ext_vector_type(8)));

union Frag   { v16bf v; uint4 u[2]; };
union Pack16 { uint4 u[2]; bf16_t h[16]; };

static __device__ __forceinline__ bf16_t f2bf16(float f) { return (bf16_t)f; }
static __device__ __forceinline__ float sigm(float x) { return 1.f / (1.f + __expf(-x)); }
static __device__ __forceinline__ float gelu_tanh(float y) {
  float z  = 0.7978845608028654f * (y + 0.044715f * y * y * y);
  float az = fabsf(z);
  float e  = __expf(-2.f * az);
  float t  = 1.f - 2.f * e / (1.f + e);     // tanh(|z|), single transcendental
  t = copysignf(t, z);
  return 0.5f * y * (1.f + t);
}

// ---------------------------------------------------------------------------
// Kernel 1: per-row fused RMSNorm stats + 30 GEMV dots + gates + 4x4 Cayley
// solve + branch = read_in . x   (one block of 256 threads per batch row)
// ---------------------------------------------------------------------------
__global__ __launch_bounds__(256)
void prep_kernel(const float* __restrict__ x,
                 const float* __restrict__ read_in_p,  const float* __restrict__ write_out_p,
                 const float* __restrict__ alpha_ri,   const float* __restrict__ alpha_wo,
                 const float* __restrict__ W_ri,       const float* __restrict__ W_wo,
                 const float* __restrict__ log_dt_c,   const float* __restrict__ log_dt_d,
                 const float* __restrict__ W_dtc,      const float* __restrict__ b_dtc,
                 const float* __restrict__ W_dtd,      const float* __restrict__ b_dtd,
                 const float* __restrict__ conserv_A,  const float* __restrict__ W_conv,
                 const float* __restrict__ diss_diag,  const float* __restrict__ W_diss,
                 bf16_t* __restrict__ branch_out,
                 float*  __restrict__ phi_out,
                 float*  __restrict__ wo_out) {
  __shared__ float xl[IDIM];           // 32 KB row stage in LDS
  __shared__ float red[31][8];
  __shared__ float ri_b[4];
  const int b = blockIdx.x;
  const int t = threadIdx.x;
  const float4* xg4 = reinterpret_cast<const float4*>(x + (size_t)b * IDIM);
  float4* xl4 = reinterpret_cast<float4*>(xl);

  float acc[31];
#pragma unroll
  for (int k = 0; k < 31; ++k) acc[k] = 0.f;

  const float4* Wri4  = reinterpret_cast<const float4*>(W_ri);
  const float4* Wwo4  = reinterpret_cast<const float4*>(W_wo);
  const float4* Wcv4  = reinterpret_cast<const float4*>(W_conv);
  const float4* Wds4  = reinterpret_cast<const float4*>(W_diss);
  const float4* Wtc4  = reinterpret_cast<const float4*>(W_dtc);
  const float4* Wtd4  = reinterpret_cast<const float4*>(W_dtd);
  const int Q = IDIM / 4;  // 2048 float4 per row

  for (int i = t; i < Q; i += 256) {
    float4 xv = xg4[i];
    xl4[i] = xv;
    acc[30] += xv.x * xv.x + xv.y * xv.y + xv.z * xv.z + xv.w * xv.w;
#pragma unroll
    for (int n = 0; n < 4; ++n) {
      float4 w = Wri4[n * Q + i];
      acc[n] += xv.x * w.x + xv.y * w.y + xv.z * w.z + xv.w * w.w;
    }
#pragma unroll
    for (int n = 0; n < 4; ++n) {
      float4 w = Wwo4[n * Q + i];
      acc[4 + n] += xv.x * w.x + xv.y * w.y + xv.z * w.z + xv.w * w.w;
    }
#pragma unroll
    for (int n = 0; n < 16; ++n) {
      float4 w = Wcv4[n * Q + i];
      acc[8 + n] += xv.x * w.x + xv.y * w.y + xv.z * w.z + xv.w * w.w;
    }
#pragma unroll
    for (int n = 0; n < 4; ++n) {
      float4 w = Wds4[n * Q + i];
      acc[24 + n] += xv.x * w.x + xv.y * w.y + xv.z * w.z + xv.w * w.w;
    }
    {
      float4 w = Wtc4[i];
      acc[28] += xv.x * w.x + xv.y * w.y + xv.z * w.z + xv.w * w.w;
      float4 v = Wtd4[i];
      acc[29] += xv.x * v.x + xv.y * v.y + xv.z * v.z + xv.w * v.w;
    }
  }

  // wave32 butterfly reduce, then cross-wave via LDS
#pragma unroll
  for (int k = 0; k < 31; ++k) {
    float v = acc[k];
    for (int o = 16; o >= 1; o >>= 1) v += __shfl_xor(v, o, 32);
    acc[k] = v;
  }
  const int lane = t & 31, wave = t >> 5;
  if (lane == 0) {
    for (int k = 0; k < 31; ++k) red[k][wave] = acc[k];
  }
  __syncthreads();

  if (t == 0) {
    float s[31];
    for (int k = 0; k < 31; ++k) {
      float v = 0.f;
      for (int w = 0; w < 8; ++w) v += red[k][w];
      s[k] = v;
    }
    const float inv_rms = rsqrtf(s[30] / (float)IDIM + REPS);
    const float a_ri = alpha_ri[0], a_wo = alpha_wo[0];
    for (int n = 0; n < 4; ++n) {
      ri_b[n] = sigm(a_ri * s[n] * inv_rms + read_in_p[n]);
      wo_out[b * 4 + n] = 2.f * sigm(a_wo * s[4 + n] * inv_rms + write_out_p[n]);
    }
    float Mm[4][4];
    for (int i2 = 0; i2 < 4; ++i2)
      for (int j = 0; j < 4; ++j)
        Mm[i2][j] = conserv_A[i2 * 4 + j] + s[8 + i2 * 4 + j] * inv_rms;
    const float dtc = DT_MIN + (DT_MAX - DT_MIN) * sigm(log_dt_c[0] + s[28] * inv_rms + b_dtc[0]);
    const float dtd = DT_MIN + (DT_MAX - DT_MIN) * sigm(log_dt_d[0] + s[29] * inv_rms + b_dtd[0]);
    float S[4][4];
    for (int i2 = 0; i2 < 4; ++i2)
      for (int j = 0; j < 4; ++j)
        S[i2][j] = dtc * 0.5f * (Mm[i2][j] - Mm[j][i2]);
    float ehD[4];
    for (int n = 0; n < 4; ++n) {
      float z = diss_diag[n] + s[24 + n] * inv_rms;
      float sp = (z > 20.f) ? z : log1pf(__expf(z));
      ehD[n] = __expf(-0.5f * dtd * sp);
    }
    // Gauss-Jordan: (I - S) Z = (I + S)
    float A4[4][4], R[4][4];
    for (int i2 = 0; i2 < 4; ++i2)
      for (int j = 0; j < 4; ++j) {
        A4[i2][j] = ((i2 == j) ? 1.f : 0.f) - S[i2][j];
        R[i2][j]  = ((i2 == j) ? 1.f : 0.f) + S[i2][j];
      }
    for (int k = 0; k < 4; ++k) {
      float inv = 1.f / A4[k][k];
      for (int j = 0; j < 4; ++j) { A4[k][j] *= inv; R[k][j] *= inv; }
      for (int r2 = 0; r2 < 4; ++r2)
        if (r2 != k) {
          float f = A4[r2][k];
          for (int j = 0; j < 4; ++j) { A4[r2][j] -= f * A4[k][j]; R[r2][j] -= f * R[k][j]; }
        }
    }
    for (int i2 = 0; i2 < 4; ++i2)
      for (int j = 0; j < 4; ++j)
        phi_out[b * 16 + i2 * 4 + j] = ehD[i2] * R[i2][j] * ehD[j];
  }
  __syncthreads();

  const float r0 = ri_b[0], r1 = ri_b[1], r2 = ri_b[2], r3 = ri_b[3];
  for (int j = t; j < EMBED; j += 256) {
    float v = r0 * xl[j] + r1 * xl[EMBED + j] + r2 * xl[2 * EMBED + j] + r3 * xl[3 * EMBED + j];
    branch_out[(size_t)b * EMBED + j] = f2bf16(v);
  }
}

// ---------------------------------------------------------------------------
// Kernel 2: one-shot W transpose + f32->bf16 convert.  src[K][N] f32 ->
// dst[N][K] bf16, 64x64 LDS tiles, both global sides coalesced b128.
// ---------------------------------------------------------------------------
__global__ __launch_bounds__(256)
void transpose_cvt_kernel(const float* __restrict__ src, bf16_t* __restrict__ dst,
                          int K, int N) {
  __shared__ bf16_t tile[64 * 72];     // pad stride 72 to spread banks
  const int t = threadIdx.x;
  const int n0 = blockIdx.x * 64;
  const int k0 = blockIdx.y * 64;
  {
    int k = t >> 2, ns = (t & 3) * 16;
    const float4* s = reinterpret_cast<const float4*>(src + (size_t)(k0 + k) * N + n0 + ns);
    Pack16 p;
#pragma unroll
    for (int q = 0; q < 4; ++q) {
      float4 f = s[q];
      p.h[q * 4 + 0] = f2bf16(f.x);
      p.h[q * 4 + 1] = f2bf16(f.y);
      p.h[q * 4 + 2] = f2bf16(f.z);
      p.h[q * 4 + 3] = f2bf16(f.w);
    }
    uint4* d = reinterpret_cast<uint4*>(&tile[k * 72 + ns]);
    d[0] = p.u[0]; d[1] = p.u[1];
  }
  __syncthreads();
  {
    int n = t >> 2, ks = (t & 3) * 16;
    Pack16 p;
#pragma unroll
    for (int j = 0; j < 16; ++j) p.h[j] = tile[(ks + j) * 72 + n];
    uint4* d = reinterpret_cast<uint4*>(dst + (size_t)(n0 + n) * K + k0 + ks);
    d[0] = p.u[0]; d[1] = p.u[1];
  }
}

// ---------------------------------------------------------------------------
// GEMM tiling shared by both matmuls: 128x128 tile, 8 waves (4x2), each wave
// 2x4 accumulators of v_wmma_f32_16x16x32_bf16, K-step 32.  A and Bt are both
// bf16 row-major with K innermost -> staging is pure uint4 copies.
// ---------------------------------------------------------------------------
#define TK  32
#define LDK 40   // padded LDS K-stride (bf16 elems)

// Kernel 3: hidden = gelu(branch @ W1), bf16 out.
__global__ __launch_bounds__(256)
void gemm1_kernel(const bf16_t* __restrict__ A,        // [BSZ][EMBED] bf16
                  const bf16_t* __restrict__ Bt,       // [DFF][EMBED] bf16 (W1^T)
                  bf16_t* __restrict__ C1) {           // [BSZ][DFF] bf16
  __shared__ bf16_t As[128 * LDK];
  __shared__ bf16_t Bs[128 * LDK];
  const int t = threadIdx.x;
  const int lane = t & 31, wave = t >> 5;
  const int wm = wave & 3, wn = wave >> 2;
  const int mTile = blockIdx.y * 128;
  const int nTile = blockIdx.x * 128;
  const int khalf = lane >> 4, lr = lane & 15;

  v8f c[2][4];
  const v8f vzero = {0.f, 0.f, 0.f, 0.f, 0.f, 0.f, 0.f, 0.f};
#pragma unroll
  for (int i = 0; i < 2; ++i)
#pragma unroll
    for (int j = 0; j < 4; ++j) c[i][j] = vzero;

  const int srow = t >> 1, skseg = (t & 1) * 16;   // staging coords (A and B)

  for (int kk = 0; kk < EMBED; kk += TK) {
    {
      const uint4* sa = reinterpret_cast<const uint4*>(
          A + (size_t)(mTile + srow) * EMBED + kk + skseg);
      const uint4* sb = reinterpret_cast<const uint4*>(
          Bt + (size_t)(nTile + srow) * EMBED + kk + skseg);
      uint4 a0 = sa[0], a1 = sa[1];
      uint4 b0 = sb[0], b1 = sb[1];
      uint4* da = reinterpret_cast<uint4*>(&As[srow * LDK + skseg]);
      uint4* db = reinterpret_cast<uint4*>(&Bs[srow * LDK + skseg]);
      da[0] = a0; da[1] = a1;
      db[0] = b0; db[1] = b1;
    }
    if (kk + TK < EMBED) {
      __builtin_prefetch(A + (size_t)(mTile + srow) * EMBED + kk + TK + skseg, 0, 3);
      __builtin_prefetch(Bt + (size_t)(nTile + srow) * EMBED + kk + TK + skseg, 0, 3);
    }
    __syncthreads();

    v16bf a[2], bfv[4];
#pragma unroll
    for (int mi = 0; mi < 2; ++mi) {       // A: lanes 0-15 K{0..7,16..23}; 16-31 K{8..15,24..31}
      int row = wm * 32 + mi * 16 + lr;
      Frag fa;
      fa.u[0] = *reinterpret_cast<const uint4*>(&As[row * LDK + khalf * 8]);
      fa.u[1] = *reinterpret_cast<const uint4*>(&As[row * LDK + 16 + khalf * 8]);
      a[mi] = fa.v;
    }
#pragma unroll
    for (int ni = 0; ni < 4; ++ni) {       // B: fixed N per lane, K{0..7,16..23} / K{8..15,24..31} like A
      int n = wn * 64 + ni * 16 + lr;
      const uint4* p = reinterpret_cast<const uint4*>(&Bs[n * LDK + khalf * 8]);
      Frag fb;
      fb.u[0] = p[0]; fb.u[1] = p[2];
      bfv[ni] = fb.v;
    }
#pragma unroll
    for (int mi = 0; mi < 2; ++mi)
#pragma unroll
      for (int ni = 0; ni < 4; ++ni)
        c[mi][ni] = __builtin_amdgcn_wmma_f32_16x16x32_bf16(
            false, a[mi], false, bfv[ni], (short)0, c[mi][ni], false, false);
    __syncthreads();
  }

  // epilogue: gelu + bf16 store.  C layout: VGPR e -> M = e + 8*(lane>>4), N = lane&15
#pragma unroll
  for (int mi = 0; mi < 2; ++mi)
#pragma unroll
    for (int ni = 0; ni < 4; ++ni)
#pragma unroll
      for (int e = 0; e < 8; ++e) {
        int row = mTile + wm * 32 + mi * 16 + e + khalf * 8;
        int col = nTile + wn * 64 + ni * 16 + lr;
        C1[(size_t)row * DFF + col] = f2bf16(gelu_tanh(c[mi][ni][e]));
      }
}

// Kernel 4: y = hidden @ W2, fused epilogue out = Phi@x_streams + write_out*y
__global__ __launch_bounds__(256)
void gemm2_kernel(const bf16_t* __restrict__ A,        // [BSZ][DFF] bf16 (hidden)
                  const bf16_t* __restrict__ Bt,       // [EMBED][DFF] bf16 (W2^T)
                  const float*  __restrict__ x,        // [BSZ][IDIM] f32
                  const float*  __restrict__ phi,      // [BSZ][16]
                  const float*  __restrict__ wo,       // [BSZ][4]
                  float* __restrict__ out) {           // [BSZ][IDIM] f32
  __shared__ bf16_t As[128 * LDK];
  __shared__ bf16_t Bs[128 * LDK];
  const int t = threadIdx.x;
  const int lane = t & 31, wave = t >> 5;
  const int wm = wave & 3, wn = wave >> 2;
  const int mTile = blockIdx.y * 128;
  const int nTile = blockIdx.x * 128;
  const int khalf = lane >> 4, lr = lane & 15;

  v8f c[2][4];
  const v8f vzero = {0.f, 0.f, 0.f, 0.f, 0.f, 0.f, 0.f, 0.f};
#pragma unroll
  for (int i = 0; i < 2; ++i)
#pragma unroll
    for (int j = 0; j < 4; ++j) c[i][j] = vzero;

  const int srow = t >> 1, skseg = (t & 1) * 16;

  for (int kk = 0; kk < DFF; kk += TK) {
    {
      const uint4* sa = reinterpret_cast<const uint4*>(
          A + (size_t)(mTile + srow) * DFF + kk + skseg);
      const uint4* sb = reinterpret_cast<const uint4*>(
          Bt + (size_t)(nTile + srow) * DFF + kk + skseg);
      uint4 a0 = sa[0], a1 = sa[1];
      uint4 b0 = sb[0], b1 = sb[1];
      uint4* da = reinterpret_cast<uint4*>(&As[srow * LDK + skseg]);
      uint4* db = reinterpret_cast<uint4*>(&Bs[srow * LDK + skseg]);
      da[0] = a0; da[1] = a1;
      db[0] = b0; db[1] = b1;
    }
    if (kk + TK < DFF) {
      __builtin_prefetch(A + (size_t)(mTile + srow) * DFF + kk + TK + skseg, 0, 3);
      __builtin_prefetch(Bt + (size_t)(nTile + srow) * DFF + kk + TK + skseg, 0, 3);
    }
    __syncthreads();

    v16bf a[2], bfv[4];
#pragma unroll
    for (int mi = 0; mi < 2; ++mi) {
      int row = wm * 32 + mi * 16 + lr;
      Frag fa;
      fa.u[0] = *reinterpret_cast<const uint4*>(&As[row * LDK + khalf * 8]);
      fa.u[1] = *reinterpret_cast<const uint4*>(&As[row * LDK + 16 + khalf * 8]);
      a[mi] = fa.v;
    }
#pragma unroll
    for (int ni = 0; ni < 4; ++ni) {
      int n = wn * 64 + ni * 16 + lr;
      const uint4* p = reinterpret_cast<const uint4*>(&Bs[n * LDK + khalf * 8]);
      Frag fb;
      fb.u[0] = p[0]; fb.u[1] = p[2];
      bfv[ni] = fb.v;
    }
#pragma unroll
    for (int mi = 0; mi < 2; ++mi)
#pragma unroll
      for (int ni = 0; ni < 4; ++ni)
        c[mi][ni] = __builtin_amdgcn_wmma_f32_16x16x32_bf16(
            false, a[mi], false, bfv[ni], (short)0, c[mi][ni], false, false);
    __syncthreads();
  }

  // fused epilogue: out[b,n,d] = sum_n' Phi[b,n,n'] x[b,n',d] + wo[b,n]*y[b,d]
#pragma unroll
  for (int mi = 0; mi < 2; ++mi)
#pragma unroll
    for (int ni = 0; ni < 4; ++ni)
#pragma unroll
      for (int e = 0; e < 8; ++e) {
        int row = mTile + wm * 32 + mi * 16 + e + khalf * 8;   // batch index b
        int col = nTile + wn * 64 + ni * 16 + lr;              // embed index d
        float y = c[mi][ni][e];
        const float* xr = x + (size_t)row * IDIM;
        const float* ph = phi + row * 16;
        const float* wr = wo + row * 4;
        float x0 = xr[col], x1 = xr[EMBED + col], x2 = xr[2 * EMBED + col], x3 = xr[3 * EMBED + col];
#pragma unroll
        for (int n = 0; n < 4; ++n) {
          float o = ph[n * 4 + 0] * x0 + ph[n * 4 + 1] * x1 +
                    ph[n * 4 + 2] * x2 + ph[n * 4 + 3] * x3 + wr[n] * y;
          out[(size_t)row * IDIM + n * EMBED + col] = o;
        }
      }
}

// ---------------------------------------------------------------------------
extern "C" void kernel_launch(void* const* d_in, const int* in_sizes, int n_in,
                              void* d_out, int out_size, void* d_ws, size_t ws_size,
                              hipStream_t stream) {
  const float* x           = (const float*)d_in[0];
  const float* read_in_p   = (const float*)d_in[1];
  const float* write_out_p = (const float*)d_in[2];
  const float* alpha_ri    = (const float*)d_in[3];
  const float* alpha_wo    = (const float*)d_in[4];
  const float* W_ri        = (const float*)d_in[5];
  const float* W_wo        = (const float*)d_in[6];
  const float* log_dt_c    = (const float*)d_in[7];
  const float* log_dt_d    = (const float*)d_in[8];
  const float* W_dtc       = (const float*)d_in[9];
  const float* b_dtc       = (const float*)d_in[10];
  const float* W_dtd       = (const float*)d_in[11];
  const float* b_dtd       = (const float*)d_in[12];
  const float* conserv_A   = (const float*)d_in[13];
  const float* W_conv      = (const float*)d_in[14];
  const float* diss_diag   = (const float*)d_in[15];
  const float* W_diss      = (const float*)d_in[16];
  const float* W1          = (const float*)d_in[17];
  const float* W2          = (const float*)d_in[18];

  char* ws = (char*)d_ws;
  size_t off = 0;
  bf16_t* branch = (bf16_t*)(ws + off); off += (size_t)BSZ * EMBED * sizeof(bf16_t);
  bf16_t* C1     = (bf16_t*)(ws + off); off += (size_t)BSZ * DFF * sizeof(bf16_t);
  bf16_t* W1t    = (bf16_t*)(ws + off); off += (size_t)DFF * EMBED * sizeof(bf16_t);
  bf16_t* W2t    = (bf16_t*)(ws + off); off += (size_t)EMBED * DFF * sizeof(bf16_t);
  float*  phi    = (float*)(ws + off);  off += (size_t)BSZ * 16 * sizeof(float);
  float*  wo     = (float*)(ws + off);  off += (size_t)BSZ * 4 * sizeof(float);
  (void)ws_size; (void)in_sizes; (void)n_in; (void)out_size;

  prep_kernel<<<BSZ, 256, 0, stream>>>(x, read_in_p, write_out_p, alpha_ri, alpha_wo,
                                       W_ri, W_wo, log_dt_c, log_dt_d, W_dtc, b_dtc,
                                       W_dtd, b_dtd, conserv_A, W_conv, diss_diag, W_diss,
                                       branch, phi, wo);

  // W1 [EMBED][DFF] -> W1t [DFF][EMBED];  W2 [DFF][EMBED] -> W2t [EMBED][DFF]
  dim3 gt1(DFF / 64, EMBED / 64);
  transpose_cvt_kernel<<<gt1, 256, 0, stream>>>(W1, W1t, EMBED, DFF);
  dim3 gt2(EMBED / 64, DFF / 64);
  transpose_cvt_kernel<<<gt2, 256, 0, stream>>>(W2, W2t, DFF, EMBED);

  dim3 g1(DFF / 128, BSZ / 128);
  gemm1_kernel<<<g1, 256, 0, stream>>>(branch, W1t, C1);

  dim3 g2(EMBED / 128, BSZ / 128);
  gemm2_kernel<<<g2, 256, 0, stream>>>(C1, W2t, x, phi, wo, (float*)d_out);
}
